// MemoryModule_28037546508335
// MI455X (gfx1250) — matrix operand order repaired and checked by reading, not compiled, not run on hardware
//
#include <hip/hip_runtime.h>

typedef __attribute__((ext_vector_type(16))) _Float16 v16h;
typedef __attribute__((ext_vector_type(8)))  _Float16 v8h;
typedef __attribute__((ext_vector_type(8)))  float    v8f;

#define N_TOT 65536
#define M_TOT 2000
#define D_TOT 256
#define ROWS  32           // z-rows per workgroup (2 WMMA row-tiles)
#define NT1   125          // M_TOT/16 n-tiles for GEMM1
#define KPAD  2016         // M_TOT padded to multiple of 32
#define KT2   63           // KPAD/32 k-tiles for GEMM2
#define SIM_STRIDE 2020    // padded fp32 row stride (bank-conflict friendly)
#define LAMB 0.002f
#define EPS_NORM 1e-10f
#define EPS_SHRINK 1e-12f

// ---- prep 1: l2-normalize memory rows -> f16 row-major [M][D] ----
__global__ void __launch_bounds__(256) prep_mnorm(const float* __restrict__ mem,
                                                  _Float16* __restrict__ mnorm) {
    int row = blockIdx.x;
    float v = mem[(size_t)row * D_TOT + threadIdx.x];
    float ss = v * v;
    #pragma unroll
    for (int o = 16; o > 0; o >>= 1) ss += __shfl_xor(ss, o, 32);
    __shared__ float red[8];
    int lane = threadIdx.x & 31, w = threadIdx.x >> 5;
    if (lane == 0) red[w] = ss;
    __syncthreads();
    float tot = red[0];
    #pragma unroll
    for (int i = 1; i < 8; i++) tot += red[i];
    float inv = 1.0f / (sqrtf(tot) + EPS_NORM);
    mnorm[(size_t)row * D_TOT + threadIdx.x] = (_Float16)(v * inv);
}

// ---- prep 2: pack memory into WMMA B-operand layout for GEMM2 ----
// B2 tile (kt,dt): element (K'=kp, N=n) -> lane = n + 16*(kp>=16), slot = kp&15
__global__ void __launch_bounds__(256) prep_packB2(const float* __restrict__ mem,
                                                   _Float16* __restrict__ b2) {
    int k = blockIdx.x;            // 0..KPAD-1
    int d = threadIdx.x;           // 0..255
    float v = (k < M_TOT) ? mem[(size_t)k * D_TOT + d] : 0.0f;
    int kt = k >> 5, kp = k & 31;
    int dt = d >> 4, n = d & 15;
    int lane = n + (kp & 16);
    int slot = kp & 15;
    b2[(((size_t)(dt * KT2 + kt) * 32 + lane) << 4) + slot] = (_Float16)v;
}

// ---- fused main kernel: 32 z-rows per workgroup, 8 wave32s ----
__global__ void __launch_bounds__(256) memmod_fused(const float* __restrict__ z,
                                                    const _Float16* __restrict__ mnorm,
                                                    const _Float16* __restrict__ b2p,
                                                    float* __restrict__ zout,
                                                    float* __restrict__ wout) {
    __shared__ float sim[ROWS * SIM_STRIDE];   // fp32 sim / exp / shrunk weights
    __shared__ _Float16 zn[ROWS * D_TOT];      // f16 normalized z rows
    __shared__ float rowScale[ROWS];

    const int lane = threadIdx.x & 31;
    const int w    = __builtin_amdgcn_readfirstlane(threadIdx.x) >> 5;  // scalar wave id
    const int r16  = lane & 15;
    const int hi   = lane >> 4;              // 0: lanes 0-15, 1: lanes 16-31

    // -------- phase 0: load + l2-normalize 4 rows per wave --------
    #pragma unroll
    for (int rr = 0; rr < 4; rr++) {
        int row = 4 * w + rr;
        size_t g = ((size_t)blockIdx.x * ROWS + row) * D_TOT;
        float v[8]; float ss = 0.0f;
        #pragma unroll
        for (int i = 0; i < 8; i++) { v[i] = z[g + lane + 32 * i]; ss += v[i] * v[i]; }
        #pragma unroll
        for (int o = 16; o > 0; o >>= 1) ss += __shfl_xor(ss, o, 32);
        float inv = 1.0f / (sqrtf(ss) + EPS_NORM);
        #pragma unroll
        for (int i = 0; i < 8; i++) zn[row * D_TOT + lane + 32 * i] = (_Float16)(v[i] * inv);
    }
    __syncthreads();

    // -------- phase 1: GEMM1  sim[32 x 2000] = z_norm @ m_norm^T --------
    // Two resident A row-tiles (rows 0-15 and 16-31); each B tile feeds 2 WMMAs.
    v16h A0[8], A1[8];
    {
        const int offh = hi << 4;    // 0 or 16 halfs
        #pragma unroll
        for (int kt = 0; kt < 8; kt++) {
            v8h l0 = *(const v8h*)&zn[r16 * D_TOT + kt * 32 + offh];
            v8h h0 = *(const v8h*)&zn[r16 * D_TOT + kt * 32 + offh + 8];
            v8h l1 = *(const v8h*)&zn[(16 + r16) * D_TOT + kt * 32 + offh];
            v8h h1 = *(const v8h*)&zn[(16 + r16) * D_TOT + kt * 32 + offh + 8];
            union { v16h v; v8h h[2]; } u0, u1;
            u0.h[0] = l0; u0.h[1] = h0; A0[kt] = u0.v;
            u1.h[0] = l1; u1.h[1] = h1; A1[kt] = u1.v;
        }
    }
    for (int j = w; j < NT1; j += 8) {
        int jp = j + 8; jp = (jp < NT1) ? jp : j;   // clamped, branch-free prefetch
        __builtin_prefetch((const void*)(mnorm + ((size_t)jp * 16 + r16) * D_TOT), 0, 3);
        v8f c0 = {}, c1 = {};
        #pragma unroll
        for (int kt = 0; kt < 8; kt++) {
            // B: lane holds column N=r16 of tile j, 16 contiguous K halfs
            const _Float16* bp = mnorm + ((size_t)(j * 16 + r16)) * D_TOT + kt * 32 + (hi << 4);
            v16h b = *(const v16h*)bp;
            c0 = __builtin_amdgcn_wmma_f32_16x16x32_f16(false, A0[kt], false, b,
                                                        (short)0, c0, false, false);
            c1 = __builtin_amdgcn_wmma_f32_16x16x32_f16(false, A1[kt], false, b,
                                                        (short)0, c1, false, false);
        }
        int col = j * 16 + r16;
        int rb  = hi << 3;
        #pragma unroll
        for (int r = 0; r < 8; r++) {
            sim[(rb + r) * SIM_STRIDE + col]        = c0[r];
            sim[(16 + rb + r) * SIM_STRIDE + col]   = c1[r];
        }
    }
    __syncthreads();

    // -------- phase 2: softmax + hard shrinkage, 4 rows per wave --------
    for (int rr = 0; rr < 4; rr++) {
        int row = 4 * w + rr;
        float* srow = sim + row * SIM_STRIDE;
        float mx = -3.0e38f;
        for (int cidx = lane; cidx < M_TOT; cidx += 32) mx = fmaxf(mx, srow[cidx]);
        #pragma unroll
        for (int o = 16; o > 0; o >>= 1) mx = fmaxf(mx, __shfl_xor(mx, o, 32));
        float s1 = 0.0f;
        for (int cidx = lane; cidx < M_TOT; cidx += 32) {
            float e = __expf(srow[cidx] - mx);
            srow[cidx] = e; s1 += e;
        }
        #pragma unroll
        for (int o = 16; o > 0; o >>= 1) s1 += __shfl_xor(s1, o, 32);
        float inv1 = 1.0f / s1;
        float s2 = 0.0f;
        for (int cidx = lane; cidx < M_TOT; cidx += 32) {
            float wv = srow[cidx] * inv1;
            float d  = wv - LAMB;
            float wp = fmaxf(d, 0.0f) * wv / (fabsf(d) + EPS_SHRINK);
            srow[cidx] = wp; s2 += wp;
        }
        #pragma unroll
        for (int o = 16; o > 0; o >>= 1) s2 += __shfl_xor(s2, o, 32);
        float scale = 1.0f / (s2 + EPS_SHRINK);
        if (lane == 0) rowScale[row] = scale;
        // write w_hat (fp32, coalesced) and stash unscaled f16 weights in place
        _Float16* arow = (_Float16*)srow;      // f16 write at 2c trails f32 read at 4c
        size_t gw = ((size_t)blockIdx.x * ROWS + row) * M_TOT;
        for (int cidx = lane; cidx < M_TOT; cidx += 32) {
            float wp = srow[cidx];
            wout[gw + cidx] = wp * scale;
            arow[cidx] = (_Float16)wp;
        }
        if (lane < KPAD - M_TOT) arow[M_TOT + lane] = (_Float16)0.0f;  // zero K padding
    }
    __syncthreads();

    // -------- phase 3: GEMM2  z_hat[32 x 256] = w_hat @ memory --------
    const _Float16* arow0 = (const _Float16*)(sim + r16 * SIM_STRIDE);
    const _Float16* arow1 = (const _Float16*)(sim + (16 + r16) * SIM_STRIDE);
    const int offh = hi << 4;
    for (int dt = w; dt < 16; dt += 8) {
        v8f c0 = {}, c1 = {};
        for (int kt = 0; kt < KT2; kt++) {
            v8h l0 = *(const v8h*)(arow0 + kt * 32 + offh);
            v8h h0 = *(const v8h*)(arow0 + kt * 32 + offh + 8);
            v8h l1 = *(const v8h*)(arow1 + kt * 32 + offh);
            v8h h1 = *(const v8h*)(arow1 + kt * 32 + offh + 8);
            union { v16h v; v8h h[2]; } u0, u1;
            u0.h[0] = l0; u0.h[1] = h0;
            u1.h[0] = l1; u1.h[1] = h1;
            v16h b = *(const v16h*)(b2p + (((size_t)(dt * KT2 + kt) * 32 + lane) << 4));
            c0 = __builtin_amdgcn_wmma_f32_16x16x32_f16(false, u0.v, false, b,
                                                        (short)0, c0, false, false);
            c1 = __builtin_amdgcn_wmma_f32_16x16x32_f16(false, u1.v, false, b,
                                                        (short)0, c1, false, false);
        }
        int col = dt * 16 + r16;
        int rb  = hi << 3;
        #pragma unroll
        for (int r = 0; r < 8; r++) {
            int row0 = rb + r, row1 = 16 + rb + r;
            zout[((size_t)blockIdx.x * ROWS + row0) * D_TOT + col] = c0[r] * rowScale[row0];
            zout[((size_t)blockIdx.x * ROWS + row1) * D_TOT + col] = c1[r] * rowScale[row1];
        }
    }
}

extern "C" void kernel_launch(void* const* d_in, const int* in_sizes, int n_in,
                              void* d_out, int out_size, void* d_ws, size_t ws_size,
                              hipStream_t stream) {
    const float* zin = (const float*)d_in[0];   // [65536, 256] fp32
    const float* mem = (const float*)d_in[1];   // [2000, 256]  fp32
    float* zout = (float*)d_out;                         // z_hat [N,D]
    float* wout = zout + (size_t)N_TOT * D_TOT;          // w_hat [N,M]
    _Float16* mnorm = (_Float16*)d_ws;                                   // 1,024,000 B
    _Float16* b2p   = (_Float16*)((char*)d_ws + (size_t)2 * 1024 * 1024); // 1,032,192 B

    prep_mnorm <<<M_TOT, 256, 0, stream>>>(mem, mnorm);
    prep_packB2<<<KPAD,  256, 0, stream>>>(mem, b2p);
    memmod_fused<<<N_TOT / ROWS, 256, 0, stream>>>(zin, mnorm, b2p, zout, wout);
}
